// SelfAttn1D_48515950575816
// MI455X (gfx1250) — compile-verified, matrix operand order checked by reading, not run on hardware
//
#include <hip/hip_runtime.h>

// MI455X / gfx1250, wave32, WMMA f16 path.
// Pipeline:
//   1) convert Wq/Wk/Wv to f16; transpose x [B,C,L] -> xh [B,L,C] f16
//   2) Qh = xh @ Wq^T + bq      [B,L,128]   (WMMA GEMM, bias on N)
//      Kh = xh @ Wk^T + bk      [B,L,128]
//      Vh = Wv @ xh^T + bv      [B,C,L]     (WMMA GEMM, bias on M)
//   3) P  = softmax(Qh @ Kh^T)  [B,L,L] f16 (fused energy+softmax, 128KB LDS strip)
//   4) y  = gamma * (P @ Vh^T)^T + x        (WMMA GEMM, fused epilogue)
//
// GEMM main loops: block tile 64x128, 8 waves (2M x 4N), each wave a 32x32
// output region = 4 WMMAs/K-step with 2x fragment reuse. Operand slabs are
// staged with GLOBAL_LOAD_ASYNC_TO_LDS_B128 (CDNA5 async engine, ASYNCcnt)
// into double-buffered LDS: no VGPR round-trip, one barrier per K-step.

typedef __attribute__((ext_vector_type(16))) _Float16 v16h;
typedef __attribute__((ext_vector_type(8)))  _Float16 v8h;
typedef __attribute__((ext_vector_type(8)))  float    v8f;
typedef __attribute__((ext_vector_type(4)))  float    v4f;

static constexpr int BATCH = 8;
static constexpr int C     = 1024;
static constexpr int L     = 2048;
static constexpr int D     = 128;
static constexpr int LDSW  = 40;  // padded row stride (f16) for 32-wide K slabs

union F16Frag { v16h v; v8h h[2]; };

// Load one 16x32 f16 fragment (A or B operand) from a row-major [rows x K]
// buffer with leading dim `ld` (elements). CDNA5 16-bit operand layout:
// lanes 0-15: K {0..7,16..23}; lanes 16-31: K {8..15,24..31}; row = lane&15.
__device__ __forceinline__ v16h load_frag16(const _Float16* base, int ld, int lane) {
  const int r  = lane & 15;
  const int kb = (lane < 16) ? 0 : 8;
  const _Float16* p = base + (size_t)r * (size_t)ld + kb;
  F16Frag f;
  f.h[0] = *(const v8h*)(p);
  f.h[1] = *(const v8h*)(p + 16);
  return f.v;
}

__device__ __forceinline__ v8f wmma16(v16h a, v16h b, v8f c) {
  return __builtin_amdgcn_wmma_f32_16x16x32_f16(false, a, false, b, (short)0, c,
                                                false, false);
}

// Async copy 16B global -> LDS (CDNA5 VGLOBAL async op, tracked by ASYNCcnt).
// Low 32 bits of a generic LDS pointer are the hardware LDS byte address
// (flat shared aperture).
__device__ __forceinline__ void async_copy_b128(const void* gaddr, void* laddr) {
  const unsigned lds_off = (unsigned)(uintptr_t)laddr;
  asm volatile("global_load_async_to_lds_b128 %0, %1, off"
               :
               : "v"(lds_off), "v"(gaddr)
               : "memory");
}

__device__ __forceinline__ void wait_async0() {
  asm volatile("s_wait_asynccnt 0" ::: "memory");
}

// ---------------------------------------------------------------- helpers ---
__global__ __launch_bounds__(256) void f32_to_f16_kernel(const float* __restrict__ s,
                                                         _Float16* __restrict__ d,
                                                         int n) {
  int i = blockIdx.x * 256 + threadIdx.x;
  if (i < n) d[i] = (_Float16)s[i];
}

// x [B,C,L] f32 -> xh [B,L,C] f16, 32x32 tiles via LDS.
__global__ __launch_bounds__(256) void x_transpose_f16_kernel(
    const float* __restrict__ x, _Float16* __restrict__ xh) {
  __shared__ _Float16 t[32][40];
  const int b  = blockIdx.z;
  const int l0 = blockIdx.x * 32;
  const int c0 = blockIdx.y * 32;
  const int tx = threadIdx.x & 31;
  const int ty = threadIdx.x >> 5;  // 0..7
  const float* xb = x + (size_t)b * C * L;
#pragma unroll
  for (int i = 0; i < 4; ++i) {
    int c = c0 + ty + i * 8;
    t[ty + i * 8][tx] = (_Float16)xb[(size_t)c * L + l0 + tx];
  }
  __syncthreads();
  _Float16* xo = xh + (size_t)b * L * C;
#pragma unroll
  for (int i = 0; i < 4; ++i) {
    int l = l0 + ty + i * 8;
    xo[(size_t)l * C + c0 + tx] = t[tx][ty + i * 8];
  }
}

// -------------------------------------------------- generic WMMA f16 GEMM ---
// D[M,N] = A[M,K] @ B[N,K]^T + bias, f16 in, f32 accumulate.
// EPILOGUE = 0: store f16 with bias[n]; 1: store f16 with bias[m];
//            2: y[b, n*ldd + m] = gamma*acc + x[b, n*ldd + m] (f32, no bias).
template <int EPILOGUE>
__global__ __launch_bounds__(256) void wmma_gemm_kernel(
    const _Float16* __restrict__ A, int lda, size_t strideA,
    const _Float16* __restrict__ Bm, int ldb, size_t strideB,
    void* __restrict__ Dp, int ldd, size_t strideD,
    const float* __restrict__ bias,          // bias (epi 0/1) or x (epi 2)
    const float* __restrict__ gamma, int K) {
  __shared__ _Float16 sA[2][64 * LDSW];
  __shared__ _Float16 sB[2][128 * LDSW];
  const int tid  = threadIdx.x;
  const int lane = tid & 31;
  const int wave = tid >> 5;
  const int wm   = wave >> 2;  // 0..1  (M direction, 32 rows each)
  const int wn   = wave & 3;   // 0..3  (N direction, 32 cols each)
  const int m0   = blockIdx.y * 64;
  const int n0   = blockIdx.x * 128;
  const int b    = blockIdx.z;
  const _Float16* Ab = A + (size_t)b * strideA;
  const _Float16* Bb = Bm + (size_t)b * strideB;

  const int row = tid >> 2;        // 0..63
  const int ch  = (tid & 3) * 8;   // 16B chunk within 32-wide K slab
  const _Float16* gA  = Ab + (size_t)(m0 + row) * lda + ch;        // 64 rows
  const _Float16* gB0 = Bb + (size_t)(n0 + row) * ldb + ch;        // rows 0..63
  const _Float16* gB1 = Bb + (size_t)(n0 + 64 + row) * ldb + ch;   // rows 64..127
  const int stOffA  = row * LDSW + ch;
  const int stOffB0 = row * LDSW + ch;
  const int stOffB1 = (64 + row) * LDSW + ch;

  // prologue: async-stage K-slab 0 into buffer 0 (no VGPR round-trip)
  async_copy_b128(gA,  &sA[0][stOffA]);
  async_copy_b128(gB0, &sB[0][stOffB0]);
  async_copy_b128(gB1, &sB[0][stOffB1]);
  wait_async0();
  __syncthreads();

  v8f acc[2][2] = {};
  int s = 0;
  for (int k0 = 0; k0 < K; k0 += 32) {
    const bool has_next = (k0 + 32) < K;
    if (has_next) {  // issue next slab's async copies; latency hides under WMMA
      async_copy_b128(gA  + k0 + 32, &sA[s ^ 1][stOffA]);
      async_copy_b128(gB0 + k0 + 32, &sB[s ^ 1][stOffB0]);
      async_copy_b128(gB1 + k0 + 32, &sB[s ^ 1][stOffB1]);
    }
    const _Float16* pA = &sA[s][(wm * 32) * LDSW];
    const _Float16* pB = &sB[s][(wn * 32) * LDSW];
    v16h af0 = load_frag16(pA, LDSW, lane);
    v16h af1 = load_frag16(pA + 16 * LDSW, LDSW, lane);
    v16h bf0 = load_frag16(pB, LDSW, lane);
    v16h bf1 = load_frag16(pB + 16 * LDSW, LDSW, lane);
    acc[0][0] = wmma16(af0, bf0, acc[0][0]);
    acc[0][1] = wmma16(af0, bf1, acc[0][1]);
    acc[1][0] = wmma16(af1, bf0, acc[1][0]);
    acc[1][1] = wmma16(af1, bf1, acc[1][1]);
    if (has_next) wait_async0();  // own async copies complete before barrier
    __syncthreads();
    s ^= 1;
  }

  const int nn = lane & 15;
  const int mh = (lane >> 4) * 8;
  if (EPILOGUE == 2) {
    const float g   = gamma[0];
    const float* xb = bias + (size_t)b * C * L;       // x, [C, L]
    float*       yb = (float*)Dp + (size_t)b * strideD;
#pragma unroll
    for (int ai = 0; ai < 2; ++ai)
#pragma unroll
      for (int bi = 0; bi < 2; ++bi)
#pragma unroll
        for (int r = 0; r < 8; ++r) {
          const int m = m0 + wm * 32 + ai * 16 + mh + r;   // position j
          const int n = n0 + wn * 32 + bi * 16 + nn;       // channel c
          const size_t idx = (size_t)n * ldd + m;          // [C, L] layout
          yb[idx] = g * acc[ai][bi][r] + xb[idx];
        }
  } else {
    _Float16* Db = (_Float16*)Dp + (size_t)b * strideD;
#pragma unroll
    for (int ai = 0; ai < 2; ++ai)
#pragma unroll
      for (int bi = 0; bi < 2; ++bi)
#pragma unroll
        for (int r = 0; r < 8; ++r) {
          const int m = m0 + wm * 32 + ai * 16 + mh + r;
          const int n = n0 + wn * 32 + bi * 16 + nn;
          const float v =
              acc[ai][bi][r] + (EPILOGUE == 1 ? bias[m] : bias[n]);
          Db[(size_t)m * ldd + n] = (_Float16)v;
        }
  }
}

// ------------------------------------- fused energy + softmax -> P (f16) ---
// One block = 16 query rows; full 16 x 2048 f32 energy strip in dynamic LDS
// (16*(2048+8)*4 = 128.5 KB; CDNA5 WGP has 320 KB). 8 waves sweep key tiles.
__global__ __launch_bounds__(256) void energy_softmax_kernel(
    const _Float16* __restrict__ Q, const _Float16* __restrict__ Kh,
    _Float16* __restrict__ P) {
  extern __shared__ float sE[];
  constexpr int ESTRIDE = L + 8;
  const int b    = blockIdx.y;
  const int j0   = blockIdx.x * 16;
  const int tid  = threadIdx.x;
  const int lane = tid & 31;
  const int wave = tid >> 5;
  const _Float16* Qb = Q + ((size_t)b * L + j0) * D;
  const _Float16* Kb = Kh + (size_t)b * L * D;

  v16h a0 = load_frag16(Qb + 0,  D, lane);
  v16h a1 = load_frag16(Qb + 32, D, lane);
  v16h a2 = load_frag16(Qb + 64, D, lane);
  v16h a3 = load_frag16(Qb + 96, D, lane);

  const int nn = lane & 15;
  const int mh = (lane >> 4) * 8;
  for (int nt = wave; nt < L / 16; nt += 8) {
    const _Float16* Kt = Kb + (size_t)nt * 16 * D;
    v8f c = {};
    c = wmma16(a0, load_frag16(Kt + 0,  D, lane), c);
    c = wmma16(a1, load_frag16(Kt + 32, D, lane), c);
    c = wmma16(a2, load_frag16(Kt + 64, D, lane), c);
    c = wmma16(a3, load_frag16(Kt + 96, D, lane), c);
#pragma unroll
    for (int r = 0; r < 8; ++r)
      sE[(mh + r) * ESTRIDE + nt * 16 + nn] = c[r];
  }
  __syncthreads();

  const int row = tid >> 4;
  const int seg = tid & 15;
  const float* rp = &sE[row * ESTRIDE + seg * 128];
  float mx = -3.402823466e38f;
  for (int i = 0; i < 128; i += 4) {
    v4f t = *(const v4f*)(rp + i);
    mx = fmaxf(mx, fmaxf(fmaxf(t[0], t[1]), fmaxf(t[2], t[3])));
  }
#pragma unroll
  for (int off = 8; off; off >>= 1) mx = fmaxf(mx, __shfl_xor(mx, off, 16));
  float sum = 0.f;
  for (int i = 0; i < 128; i += 4) {
    v4f t = *(const v4f*)(rp + i);
    sum += __expf(t[0] - mx) + __expf(t[1] - mx) +
           __expf(t[2] - mx) + __expf(t[3] - mx);
  }
#pragma unroll
  for (int off = 8; off; off >>= 1) sum += __shfl_xor(sum, off, 16);
  const float inv = 1.f / sum;

  _Float16* pout = P + ((size_t)b * L + j0 + row) * L + seg * 128;
  for (int i = 0; i < 128; i += 8) {
    v8h o;
#pragma unroll
    for (int u = 0; u < 8; ++u)
      o[u] = (_Float16)(__expf(rp[i + u] - mx) * inv);
    *(v8h*)(pout + i) = o;
  }
}

// ------------------------------------------------------------------ driver ---
extern "C" void kernel_launch(void* const* d_in, const int* in_sizes, int n_in,
                              void* d_out, int out_size, void* d_ws, size_t ws_size,
                              hipStream_t stream) {
  const float* x     = (const float*)d_in[0];
  const float* Wq    = (const float*)d_in[1];
  const float* bq    = (const float*)d_in[2];
  const float* Wk    = (const float*)d_in[3];
  const float* bk    = (const float*)d_in[4];
  const float* Wv    = (const float*)d_in[5];
  const float* bv    = (const float*)d_in[6];
  const float* gamma = (const float*)d_in[7];
  float*       y     = (float*)d_out;

  char* ws = (char*)d_ws;
  _Float16* xh  = (_Float16*)ws;  ws += (size_t)BATCH * L * C * 2;  // 33.5 MB
  _Float16* wqh = (_Float16*)ws;  ws += (size_t)D * C * 2;
  _Float16* wkh = (_Float16*)ws;  ws += (size_t)D * C * 2;
  _Float16* wvh = (_Float16*)ws;  ws += (size_t)C * C * 2;
  _Float16* qh  = (_Float16*)ws;  ws += (size_t)BATCH * L * D * 2;
  _Float16* kh  = (_Float16*)ws;  ws += (size_t)BATCH * L * D * 2;
  _Float16* vh  = (_Float16*)ws;  ws += (size_t)BATCH * C * L * 2;  // 33.5 MB
  _Float16* ph  = (_Float16*)ws;  ws += (size_t)BATCH * L * L * 2;  // 64 MB

  // 1) weight conversion + x transpose/convert
  f32_to_f16_kernel<<<(D * C + 255) / 256, 256, 0, stream>>>(Wq, wqh, D * C);
  f32_to_f16_kernel<<<(D * C + 255) / 256, 256, 0, stream>>>(Wk, wkh, D * C);
  f32_to_f16_kernel<<<(C * C + 255) / 256, 256, 0, stream>>>(Wv, wvh, C * C);
  x_transpose_f16_kernel<<<dim3(L / 32, C / 32, BATCH), 256, 0, stream>>>(x, xh);

  // 2) projections (block tile 64x128)
  wmma_gemm_kernel<0><<<dim3(D / 128, L / 64, BATCH), 256, 0, stream>>>(
      xh, C, (size_t)L * C, wqh, C, 0, qh, D, (size_t)L * D, bq, nullptr, C);
  wmma_gemm_kernel<0><<<dim3(D / 128, L / 64, BATCH), 256, 0, stream>>>(
      xh, C, (size_t)L * C, wkh, C, 0, kh, D, (size_t)L * D, bk, nullptr, C);
  wmma_gemm_kernel<1><<<dim3(L / 128, C / 64, BATCH), 256, 0, stream>>>(
      wvh, C, 0, xh, C, (size_t)L * C, vh, L, (size_t)C * L, bv, nullptr, C);

  // 3) energy + softmax (128.5 KB dynamic LDS per block)
  const size_t lds_bytes = (size_t)16 * (L + 8) * sizeof(float);
  energy_softmax_kernel<<<dim3(L / 16, BATCH), 256, lds_bytes, stream>>>(qh, kh, ph);

  // 4) output GEMM: M=L (j), N=C (c), K=L; fused y = gamma*out + x
  wmma_gemm_kernel<2><<<dim3(C / 128, L / 64, BATCH), 256, 0, stream>>>(
      ph, L, (size_t)L * L, vh, L, (size_t)C * L, y, L, (size_t)C * L,
      x, gamma, L);
}
